// get_model_68702296867336
// MI455X (gfx1250) — matrix operand order repaired and checked by reading, not compiled, not run on hardware
//
#include <hip/hip_runtime.h>
#include <hip/hip_bf16.h>

// ---------------------------------------------------------------------------
// MI455X (gfx1250, wave32) implementation of the point-cloud seg reference.
//
// Input order assumption (setup_inputs dict insertion order, params flattened
// in _init_params insertion order):
//   d_in[0]  xyz (8*6*2048 f32)       d_in[1]  cls_label (unused)
//   d_in[2..35] params interleaved (weight, bias):
//     pw1 pb1 pw2 pb2 pw3 pb3 aw1 ab1 aw2 ab2 aw3 ab3 bw1 bb1 bw2 bb2 bw3 bb3
//     c1 cb1 c2 cb2 c3 cb3 c4 cb4 c5 cb5 c6 cb6 c7 cb7 c8 cb8
//   d_in[36] a_s (512 f32)
// Output: (8,2048,50) f32 log-softmax.
// Workspace requirement: ~461 MB.
// ---------------------------------------------------------------------------

#define B_    8
#define N_    2048
#define NB_   512
#define NN_   1536
#define KNN_  32
#define CH_   6
#define NCLS_ 50

typedef __attribute__((ext_vector_type(16))) _Float16 v16h;
typedef __attribute__((ext_vector_type(8)))  float    v8f;

// ---------------------------------------------------------------------------
// WMMA GEMM:  Y[M x O] = act( X[M x K] @ W[O x K]^T + bias[O] )
//
// One wave computes a 16x64 strip: 4 accumulator tiles (v_wmma_f32_16x16x32_f16)
// sharing one A fragment per K-step.  Block = 256 threads = 8 waves
// -> 128 rows x 64 cols per block.  grid = (ceil(O/64), M/128, B).
//
// Constraints exploited (all true for every layer in this net):
//   * M % 128 == 0                      -> no row guards anywhere
//   * K % 32 == 0  OR  K < 32           -> main loop unguarded float4 loads,
//                                          single clamped-select tail for K=6/12
//   * ragged O handled by clamping B row pointers (garbage cols never stored)
// ---------------------------------------------------------------------------
template <bool RELU>
__global__ void wmma_gemm_kernel(const float* __restrict__ X,
                                 const float* __restrict__ W,
                                 const float* __restrict__ bias,
                                 float* __restrict__ Y,
                                 int M, int K, int O)
{
    const int b = blockIdx.z;
    X += (long long)b * M * K;
    Y += (long long)b * M * O;

    const int wave   = threadIdx.x >> 5;
    const int lane   = threadIdx.x & 31;
    const int mbase  = blockIdx.y * 128 + wave * 16;   // always < M (M%128==0)
    const int obase0 = blockIdx.x * 64;

    const int  hl = lane & 15;
    const bool hi = lane >= 16;

    const int m = mbase + hl;                          // A row (always valid)
    const float* __restrict__ xrow = X + (long long)m * K;

    // B row pointers for the 4 tiles, clamped in-bounds (ragged O safe).
    const float* __restrict__ wrow[4];
    int ocol[4];
    #pragma unroll
    for (int t = 0; t < 4; ++t) {
        const int o = obase0 + 16 * t + hl;
        ocol[t] = o;
        const int oc = (o < O) ? o : (O - 1);
        wrow[t] = W + (long long)oc * K;
    }

    v8f acc0 = {}, acc1 = {}, acc2 = {}, acc3 = {};

    const int aoff = hi ? 8 : 0;                       // A sub-lane K offset
    const int boff = hi ? 8 : 0;                       // B sub-lane K offset
    const int kFull = K & ~31;

    for (int kb = 0; kb < kFull; kb += 32) {
        // ---- A fragment: 16x32 f16, unconditional float4 loads ----
        const float* ax = xrow + kb;
        const float4 a0 = *(const float4*)(ax + aoff);
        const float4 a1 = *(const float4*)(ax + aoff + 4);
        const float4 a2 = *(const float4*)(ax + 16 + aoff);
        const float4 a3 = *(const float4*)(ax + 16 + aoff + 4);
        v16h a;
        a[0]=(_Float16)a0.x;  a[1]=(_Float16)a0.y;  a[2]=(_Float16)a0.z;  a[3]=(_Float16)a0.w;
        a[4]=(_Float16)a1.x;  a[5]=(_Float16)a1.y;  a[6]=(_Float16)a1.z;  a[7]=(_Float16)a1.w;
        a[8]=(_Float16)a2.x;  a[9]=(_Float16)a2.y;  a[10]=(_Float16)a2.z; a[11]=(_Float16)a2.w;
        a[12]=(_Float16)a3.x; a[13]=(_Float16)a3.y; a[14]=(_Float16)a3.z; a[15]=(_Float16)a3.w;

        // ---- 4 B fragments + 4 WMMAs ----
        #pragma unroll
        for (int t = 0; t < 4; ++t) {
            const float* wx = wrow[t] + kb + boff;
            const float4 b0 = *(const float4*)(wx + 0);
            const float4 b1 = *(const float4*)(wx + 4);
            const float4 b2 = *(const float4*)(wx + 16);
            const float4 b3 = *(const float4*)(wx + 20);
            v16h bm;
            bm[0]=(_Float16)b0.x;  bm[1]=(_Float16)b0.y;  bm[2]=(_Float16)b0.z;  bm[3]=(_Float16)b0.w;
            bm[4]=(_Float16)b1.x;  bm[5]=(_Float16)b1.y;  bm[6]=(_Float16)b1.z;  bm[7]=(_Float16)b1.w;
            bm[8]=(_Float16)b2.x;  bm[9]=(_Float16)b2.y;  bm[10]=(_Float16)b2.z; bm[11]=(_Float16)b2.w;
            bm[12]=(_Float16)b3.x; bm[13]=(_Float16)b3.y; bm[14]=(_Float16)b3.z; bm[15]=(_Float16)b3.w;
            if      (t == 0) acc0 = __builtin_amdgcn_wmma_f32_16x16x32_f16(false, a, false, bm, (short)0, acc0, false, false);
            else if (t == 1) acc1 = __builtin_amdgcn_wmma_f32_16x16x32_f16(false, a, false, bm, (short)0, acc1, false, false);
            else if (t == 2) acc2 = __builtin_amdgcn_wmma_f32_16x16x32_f16(false, a, false, bm, (short)0, acc2, false, false);
            else             acc3 = __builtin_amdgcn_wmma_f32_16x16x32_f16(false, a, false, bm, (short)0, acc3, false, false);
        }
    }

    // ---- K tail (only K=6 / K=12 reach this): clamped loads + value select,
    //      no exec-mask divergence.
    if (kFull < K) {
        const int kb = kFull;
        v16h a;
        #pragma unroll
        for (int e = 0; e < 8; ++e) {
            const int k0 = kb + aoff + e;
            const int k1 = kb + 16 + aoff + e;
            const float v0 = xrow[(k0 < K) ? k0 : (K - 1)];
            const float v1 = xrow[(k1 < K) ? k1 : (K - 1)];
            a[e]     = (k0 < K) ? (_Float16)v0 : (_Float16)0.f;
            a[e + 8] = (k1 < K) ? (_Float16)v1 : (_Float16)0.f;
        }
        #pragma unroll
        for (int t = 0; t < 4; ++t) {
            v16h bm;
            #pragma unroll
            for (int e = 0; e < 16; ++e) {
                const int k = kb + boff + e + (e >= 8 ? 8 : 0);
                const float v = wrow[t][(k < K) ? k : (K - 1)];
                bm[e] = (k < K) ? (_Float16)v : (_Float16)0.f;
            }
            if      (t == 0) acc0 = __builtin_amdgcn_wmma_f32_16x16x32_f16(false, a, false, bm, (short)0, acc0, false, false);
            else if (t == 1) acc1 = __builtin_amdgcn_wmma_f32_16x16x32_f16(false, a, false, bm, (short)0, acc1, false, false);
            else if (t == 2) acc2 = __builtin_amdgcn_wmma_f32_16x16x32_f16(false, a, false, bm, (short)0, acc2, false, false);
            else             acc3 = __builtin_amdgcn_wmma_f32_16x16x32_f16(false, a, false, bm, (short)0, acc3, false, false);
        }
    }

    // ---- store: C/D layout -> VGPR r = row mbase + r + (hi?8:0), col = ocol[t]
    const int rrow = mbase + (hi ? 8 : 0);
    #pragma unroll
    for (int t = 0; t < 4; ++t) {
        const int n = ocol[t];
        if (n < O) {
            const float bv = bias[n];
            const v8f acc = (t == 0) ? acc0 : (t == 1) ? acc1 : (t == 2) ? acc2 : acc3;
            #pragma unroll
            for (int r = 0; r < 8; ++r) {
                float v = acc[r] + bv;
                if (RELU) v = fmaxf(v, 0.0f);
                Y[(long long)(rrow + r) * O + n] = v;
            }
        }
    }
}

// ---------------------------------------------------------------------------
// Transpose xyz (B,6,N) -> ptsT (B,N,6) and sq = |xyz[:3]|^2
// ---------------------------------------------------------------------------
__global__ void prep_kernel(const float* __restrict__ xyz,
                            float* __restrict__ ptsT, float* __restrict__ sq)
{
    const int b = blockIdx.z;
    const int i = blockIdx.x * blockDim.x + threadIdx.x;
    if (i >= N_) return;
    const float* src = xyz + (long long)b * CH_ * N_;
    float* dst = ptsT + ((long long)b * N_ + i) * CH_;
    float s = 0.f;
    #pragma unroll
    for (int c = 0; c < CH_; ++c) {
        const float v = src[c * N_ + i];
        dst[c] = v;
        if (c < 3) s += v * v;
    }
    sq[(long long)b * N_ + i] = s;
}

// ---------------------------------------------------------------------------
// Per-point 32-NN (incl. self) -> offset = || mean(neigh) - xyz ||
// ---------------------------------------------------------------------------
__global__ void knn_offset_kernel(const float* __restrict__ ptsT,
                                  const float* __restrict__ sq,
                                  float* __restrict__ offs)
{
    __shared__ float dist[N_];
    __shared__ float rmin[256];
    __shared__ int   rarg[256];
    const int b = blockIdx.z;
    const int i = blockIdx.x;
    const float* P  = ptsT + (long long)b * N_ * CH_;
    const float* SQ = sq   + (long long)b * N_;
    const float xi = P[i * CH_ + 0], yi = P[i * CH_ + 1], zi = P[i * CH_ + 2];
    const float sqi = SQ[i];
    for (int j = threadIdx.x; j < N_; j += blockDim.x)
        dist[j] = sqi + SQ[j]
                - 2.f * (xi * P[j * CH_ + 0] + yi * P[j * CH_ + 1] + zi * P[j * CH_ + 2]);
    __syncthreads();

    float mx = 0.f, my = 0.f, mz = 0.f;
    for (int t = 0; t < KNN_; ++t) {
        float best = 1e30f; int barg = 0;
        for (int j = threadIdx.x; j < N_; j += blockDim.x)
            if (dist[j] < best) { best = dist[j]; barg = j; }
        rmin[threadIdx.x] = best; rarg[threadIdx.x] = barg;
        __syncthreads();
        for (int s = blockDim.x >> 1; s > 0; s >>= 1) {
            if (threadIdx.x < s && rmin[threadIdx.x + s] < rmin[threadIdx.x]) {
                rmin[threadIdx.x] = rmin[threadIdx.x + s];
                rarg[threadIdx.x] = rarg[threadIdx.x + s];
            }
            __syncthreads();
        }
        const int sel = rarg[0];
        mx += P[sel * CH_ + 0]; my += P[sel * CH_ + 1]; mz += P[sel * CH_ + 2];
        if (threadIdx.x == 0) dist[sel] = 1e30f;
        __syncthreads();
    }
    if (threadIdx.x == 0) {
        mx *= (1.f / KNN_); my *= (1.f / KNN_); mz *= (1.f / KNN_);
        const float dx = mx - xi, dy = my - yi, dz = mz - zi;
        offs[(long long)b * N_ + i] = sqrtf(dx * dx + dy * dy + dz * dz);
    }
}

// ---------------------------------------------------------------------------
// Per-batch bitonic sort of (offset desc, index) -> order permutation.
// ---------------------------------------------------------------------------
__global__ void sort_kernel(const float* __restrict__ offs, int* __restrict__ order)
{
    __shared__ float key[N_];
    __shared__ int   val[N_];
    const int b = blockIdx.x;
    for (int i = threadIdx.x; i < N_; i += blockDim.x) {
        key[i] = offs[(long long)b * N_ + i];
        val[i] = i;
    }
    __syncthreads();
    for (int k = 2; k <= N_; k <<= 1) {
        for (int j = k >> 1; j > 0; j >>= 1) {
            for (int i = threadIdx.x; i < N_; i += blockDim.x) {
                const int ixj = i ^ j;
                if (ixj > i) {
                    const bool descBlock = ((i & k) == 0);
                    const bool swap = descBlock ? (key[i] < key[ixj])
                                                : (key[i] > key[ixj]);
                    if (swap) {
                        const float tk = key[i]; key[i] = key[ixj]; key[ixj] = tk;
                        const int   tv = val[i]; val[i] = val[ixj]; val[ixj] = tv;
                    }
                }
            }
            __syncthreads();
        }
    }
    for (int i = threadIdx.x; i < N_; i += blockDim.x)
        order[(long long)b * N_ + i] = val[i];
}

// ---------------------------------------------------------------------------
// Gather boundary / non-boundary point sets.
// ---------------------------------------------------------------------------
__global__ void gather_pts_kernel(const float* __restrict__ ptsT,
                                  const int* __restrict__ order,
                                  float* __restrict__ bpts, float* __restrict__ npts)
{
    const int b = blockIdx.z;
    const int i = blockIdx.x * blockDim.x + threadIdx.x;
    if (i >= N_) return;
    const int src = order[(long long)b * N_ + i];
    const float* p = ptsT + ((long long)b * N_ + src) * CH_;
    float* d = (i < NB_) ? bpts + ((long long)b * NB_ + i) * CH_
                         : npts + ((long long)b * NN_ + (i - NB_)) * CH_;
    #pragma unroll
    for (int c = 0; c < CH_; ++c) d[c] = p[c];
}

// ---------------------------------------------------------------------------
// Attention score: s_i = af[i] . a_s
// ---------------------------------------------------------------------------
__global__ void attn_score_kernel(const float* __restrict__ af,
                                  const float* __restrict__ a_s,
                                  float* __restrict__ s)
{
    const int b = blockIdx.z;
    const int i = blockIdx.x * blockDim.x + threadIdx.x;
    if (i >= N_) return;
    const float* row = af + ((long long)b * N_ + i) * 512;
    float acc = 0.f;
    for (int d = 0; d < 512; ++d) acc += row[d] * a_s[d];
    s[(long long)b * N_ + i] = acc;
}

// ---------------------------------------------------------------------------
// softmax over N + g = attn @ af (512-dim). One 512-thread block per batch.
// ---------------------------------------------------------------------------
__global__ void softmax_g_kernel(const float* __restrict__ s,
                                 const float* __restrict__ af,
                                 float* __restrict__ g)
{
    __shared__ float ev[N_];
    __shared__ float red[512];
    const int b = blockIdx.x;
    const int t = threadIdx.x;
    float mx = -1e30f;
    for (int i = t; i < N_; i += blockDim.x) mx = fmaxf(mx, s[(long long)b * N_ + i]);
    red[t] = mx; __syncthreads();
    for (int st = blockDim.x >> 1; st > 0; st >>= 1) {
        if (t < st) red[t] = fmaxf(red[t], red[t + st]);
        __syncthreads();
    }
    mx = red[0]; __syncthreads();
    float sum = 0.f;
    for (int i = t; i < N_; i += blockDim.x) {
        const float v = expf(s[(long long)b * N_ + i] - mx);
        ev[i] = v; sum += v;
    }
    red[t] = sum; __syncthreads();
    for (int st = blockDim.x >> 1; st > 0; st >>= 1) {
        if (t < st) red[t] += red[t + st];
        __syncthreads();
    }
    const float inv = 1.f / red[0];
    float acc = 0.f;
    for (int i = 0; i < N_; ++i)
        acc += ev[i] * af[((long long)b * N_ + i) * 512 + t];
    g[(long long)b * 512 + t] = acc * inv;
}

__global__ void add_g_kernel(float* __restrict__ af, const float* __restrict__ g)
{
    const int b = blockIdx.z;
    const long long idx = (long long)blockIdx.x * blockDim.x + threadIdx.x;
    if (idx >= (long long)N_ * 512) return;
    const int d = (int)(idx & 511);
    af[(long long)b * N_ * 512 + idx] += g[(long long)b * 512 + d];
}

// ---------------------------------------------------------------------------
// Boundary knn: 32-NN of each boundary point over all N points -> indices.
// ---------------------------------------------------------------------------
__global__ void bknn_kernel(const float* __restrict__ ptsT,
                            const float* __restrict__ sq,
                            const float* __restrict__ bpts,
                            int* __restrict__ bknn)
{
    __shared__ float dist[N_];
    __shared__ float rmin[256];
    __shared__ int   rarg[256];
    const int b  = blockIdx.z;
    const int bi = blockIdx.x;
    const float* c6 = bpts + ((long long)b * NB_ + bi) * CH_;
    const float xi = c6[0], yi = c6[1], zi = c6[2];
    const float sqi = xi * xi + yi * yi + zi * zi;
    const float* P  = ptsT + (long long)b * N_ * CH_;
    const float* SQ = sq   + (long long)b * N_;
    for (int j = threadIdx.x; j < N_; j += blockDim.x)
        dist[j] = sqi + SQ[j]
                - 2.f * (xi * P[j * CH_ + 0] + yi * P[j * CH_ + 1] + zi * P[j * CH_ + 2]);
    __syncthreads();
    for (int t = 0; t < KNN_; ++t) {
        float best = 1e30f; int barg = 0;
        for (int j = threadIdx.x; j < N_; j += blockDim.x)
            if (dist[j] < best) { best = dist[j]; barg = j; }
        rmin[threadIdx.x] = best; rarg[threadIdx.x] = barg;
        __syncthreads();
        for (int s = blockDim.x >> 1; s > 0; s >>= 1) {
            if (threadIdx.x < s && rmin[threadIdx.x + s] < rmin[threadIdx.x]) {
                rmin[threadIdx.x] = rmin[threadIdx.x + s];
                rarg[threadIdx.x] = rarg[threadIdx.x + s];
            }
            __syncthreads();
        }
        const int sel = rarg[0];
        if (threadIdx.x == 0) {
            bknn[(((long long)b * NB_ + bi) * KNN_) + t] = sel;
            dist[sel] = 1e30f;
        }
        __syncthreads();
    }
}

// ---------------------------------------------------------------------------
// edge rows: [center(6), neigh-center(6)] per (bi, kk)
// ---------------------------------------------------------------------------
__global__ void edge_kernel(const float* __restrict__ ptsT,
                            const float* __restrict__ bpts,
                            const int* __restrict__ bknn,
                            float* __restrict__ edge)
{
    const int b = blockIdx.z;
    const int r = blockIdx.x * blockDim.x + threadIdx.x;  // bi*32 + kk
    if (r >= NB_ * KNN_) return;
    const int bi = r >> 5;
    const float* c6 = bpts + ((long long)b * NB_ + bi) * CH_;
    const int nidx = bknn[(long long)b * NB_ * KNN_ + r];
    const float* n6 = ptsT + ((long long)b * N_ + nidx) * CH_;
    float* e = edge + ((long long)b * NB_ * KNN_ + r) * (2 * CH_);
    #pragma unroll
    for (int c = 0; c < CH_; ++c) {
        e[c]        = c6[c];
        e[CH_ + c]  = n6[c] - c6[c];
    }
}

// ---------------------------------------------------------------------------
// maxpool over the 32 neighbors
// ---------------------------------------------------------------------------
__global__ void maxpool_kernel(const float* __restrict__ e3, float* __restrict__ bfeat)
{
    const int b = blockIdx.z;
    const int idx = blockIdx.x * blockDim.x + threadIdx.x;   // bi*256 + c
    if (idx >= NB_ * 256) return;
    const int bi = idx >> 8, c = idx & 255;
    float m = -1e30f;
    for (int kk = 0; kk < KNN_; ++kk)
        m = fmaxf(m, e3[(((long long)b * NB_ + bi) * KNN_ + kk) * 256 + c]);
    bfeat[((long long)b * NB_ + bi) * 256 + c] = m;
}

// ---------------------------------------------------------------------------
// concats
// ---------------------------------------------------------------------------
__global__ void concat_b_kernel(const float* __restrict__ bfeat,
                                const float* __restrict__ gfeat,
                                const int* __restrict__ order,
                                float* __restrict__ bcat)
{
    const int b = blockIdx.z;
    const int idx = blockIdx.x * blockDim.x + threadIdx.x;  // i*768 + c
    if (idx >= NB_ * 768) return;
    const int i = idx / 768, c = idx % 768;
    float v;
    if (c < 256) v = bfeat[((long long)b * NB_ + i) * 256 + c];
    else {
        const int gi = order[(long long)b * N_ + i];
        v = gfeat[((long long)b * N_ + gi) * 512 + (c - 256)];
    }
    bcat[((long long)b * NB_ + i) * 768 + c] = v;
}

__global__ void concat_n_kernel(const float* __restrict__ nb3,
                                const float* __restrict__ gfeat,
                                const int* __restrict__ order,
                                float* __restrict__ ncat)
{
    const int b = blockIdx.z;
    const long long idx = (long long)blockIdx.x * blockDim.x + threadIdx.x; // i*1024+c
    if (idx >= (long long)NN_ * 1024) return;
    const int i = (int)(idx >> 10), c = (int)(idx & 1023);
    float v;
    if (c < 512) v = nb3[((long long)b * NN_ + i) * 512 + c];
    else {
        const int gi = order[(long long)b * N_ + NB_ + i];
        v = gfeat[((long long)b * N_ + gi) * 512 + (c - 512)];
    }
    ncat[((long long)b * NN_ + i) * 1024 + c] = v;
}

// ---------------------------------------------------------------------------
// log_softmax + scatter to out[order[i]]
// ---------------------------------------------------------------------------
__global__ void logsoftmax_scatter_kernel(const float* __restrict__ blog,
                                          const float* __restrict__ nlog,
                                          const int* __restrict__ order,
                                          float* __restrict__ out)
{
    const int b = blockIdx.z;
    const int i = blockIdx.x * blockDim.x + threadIdx.x;
    if (i >= N_) return;
    const float* src = (i < NB_)
        ? blog + ((long long)b * NB_ + i) * NCLS_
        : nlog + ((long long)b * NN_ + (i - NB_)) * NCLS_;
    const int dst = order[(long long)b * N_ + i];
    float mx = -1e30f;
    for (int c = 0; c < NCLS_; ++c) mx = fmaxf(mx, src[c]);
    float s = 0.f;
    for (int c = 0; c < NCLS_; ++c) s += expf(src[c] - mx);
    const float lse = mx + logf(s);
    float* o = out + ((long long)b * N_ + dst) * NCLS_;
    for (int c = 0; c < NCLS_; ++c) o[c] = src[c] - lse;
}

// ---------------------------------------------------------------------------
// host launcher
// ---------------------------------------------------------------------------
static void launch_gemm(const float* X, const float* W, const float* bias, float* Y,
                        int M, int K, int O, bool relu, hipStream_t st)
{
    // M must be a multiple of 128 (true for all layers here).
    dim3 grid((O + 63) / 64, M / 128, B_);
    if (relu)
        wmma_gemm_kernel<true><<<grid, 256, 0, st>>>(X, W, bias, Y, M, K, O);
    else
        wmma_gemm_kernel<false><<<grid, 256, 0, st>>>(X, W, bias, Y, M, K, O);
}

extern "C" void kernel_launch(void* const* d_in, const int* in_sizes, int n_in,
                              void* d_out, int out_size, void* d_ws, size_t ws_size,
                              hipStream_t stream)
{
    const float* xyz = (const float*)d_in[0];
    const float *pw1 = (const float*)d_in[2],  *pb1 = (const float*)d_in[3];
    const float *pw2 = (const float*)d_in[4],  *pb2 = (const float*)d_in[5];
    const float *pw3 = (const float*)d_in[6],  *pb3 = (const float*)d_in[7];
    const float *aw1 = (const float*)d_in[8],  *ab1 = (const float*)d_in[9];
    const float *aw2 = (const float*)d_in[10], *ab2 = (const float*)d_in[11];
    const float *aw3 = (const float*)d_in[12], *ab3 = (const float*)d_in[13];
    const float *bw1 = (const float*)d_in[14], *bb1 = (const float*)d_in[15];
    const float *bw2 = (const float*)d_in[16], *bb2 = (const float*)d_in[17];
    const float *bw3 = (const float*)d_in[18], *bb3 = (const float*)d_in[19];
    const float *c1  = (const float*)d_in[20], *cb1 = (const float*)d_in[21];
    const float *c2  = (const float*)d_in[22], *cb2 = (const float*)d_in[23];
    const float *c3  = (const float*)d_in[24], *cb3 = (const float*)d_in[25];
    const float *c4  = (const float*)d_in[26], *cb4 = (const float*)d_in[27];
    const float *c5  = (const float*)d_in[28], *cb5 = (const float*)d_in[29];
    const float *c6  = (const float*)d_in[30], *cb6 = (const float*)d_in[31];
    const float *c7  = (const float*)d_in[32], *cb7 = (const float*)d_in[33];
    const float *c8  = (const float*)d_in[34], *cb8 = (const float*)d_in[35];
    const float *a_s = (const float*)d_in[36];
    float* out = (float*)d_out;

    // workspace arena (~461 MB)
    char* w = (char*)d_ws;
    size_t off = 0;
    auto alloc = [&](size_t elems, size_t esz) -> void* {
        off = (off + 255) & ~(size_t)255;
        void* p = w + off;
        off += elems * esz;
        return p;
    };
    float* ptsT  = (float*)alloc((size_t)B_ * N_ * CH_, 4);
    float* sq    = (float*)alloc((size_t)B_ * N_, 4);
    float* offs  = (float*)alloc((size_t)B_ * N_, 4);
    int*   order = (int*)  alloc((size_t)B_ * N_, 4);
    float* afh1  = (float*)alloc((size_t)B_ * N_ * 64, 4);
    float* afh2  = (float*)alloc((size_t)B_ * N_ * 256, 4);
    float* af    = (float*)alloc((size_t)B_ * N_ * 512, 4);
    float* attnS = (float*)alloc((size_t)B_ * N_, 4);
    float* gvec  = (float*)alloc((size_t)B_ * 512, 4);
    float* npts  = (float*)alloc((size_t)B_ * NN_ * CH_, 4);
    float* nb1   = (float*)alloc((size_t)B_ * NN_ * 64, 4);
    float* nb2   = (float*)alloc((size_t)B_ * NN_ * 128, 4);
    float* nb3   = (float*)alloc((size_t)B_ * NN_ * 512, 4);
    float* bpts  = (float*)alloc((size_t)B_ * NB_ * CH_, 4);
    int*   bknn  = (int*)  alloc((size_t)B_ * NB_ * KNN_, 4);
    float* edge  = (float*)alloc((size_t)B_ * NB_ * KNN_ * 2 * CH_, 4);
    float* e1    = (float*)alloc((size_t)B_ * NB_ * KNN_ * 64, 4);
    float* e2    = (float*)alloc((size_t)B_ * NB_ * KNN_ * 128, 4);
    float* e3    = (float*)alloc((size_t)B_ * NB_ * KNN_ * 256, 4);
    float* bfeat = (float*)alloc((size_t)B_ * NB_ * 256, 4);
    float* bcat  = (float*)alloc((size_t)B_ * NB_ * 768, 4);
    float* ncat  = (float*)alloc((size_t)B_ * NN_ * 1024, 4);
    float* bh1   = (float*)alloc((size_t)B_ * NB_ * 512, 4);
    float* bh2   = (float*)alloc((size_t)B_ * NB_ * 256, 4);
    float* bh3   = (float*)alloc((size_t)B_ * NB_ * 128, 4);
    float* blog  = (float*)alloc((size_t)B_ * NB_ * NCLS_, 4);
    float* nh1   = (float*)alloc((size_t)B_ * NN_ * 512, 4);
    float* nh2   = (float*)alloc((size_t)B_ * NN_ * 256, 4);
    float* nh3   = (float*)alloc((size_t)B_ * NN_ * 128, 4);
    float* nlog  = (float*)alloc((size_t)B_ * NN_ * NCLS_, 4);
    (void)ws_size; (void)in_sizes; (void)n_in; (void)out_size;

    // ---- geometry / selection ----
    prep_kernel<<<dim3((N_ + 255) / 256, 1, B_), 256, 0, stream>>>(xyz, ptsT, sq);
    knn_offset_kernel<<<dim3(N_, 1, B_), 256, 0, stream>>>(ptsT, sq, offs);
    sort_kernel<<<B_, 1024, 0, stream>>>(offs, order);
    gather_pts_kernel<<<dim3((N_ + 255) / 256, 1, B_), 256, 0, stream>>>(ptsT, order, bpts, npts);

    // ---- attention branch (af) ----
    launch_gemm(ptsT, aw1, ab1, afh1, N_, CH_, 64, true, stream);
    launch_gemm(afh1, aw2, ab2, afh2, N_, 64, 256, true, stream);
    launch_gemm(afh2, aw3, ab3, af,   N_, 256, 512, true, stream);
    attn_score_kernel<<<dim3((N_ + 255) / 256, 1, B_), 256, 0, stream>>>(af, a_s, attnS);
    softmax_g_kernel<<<B_, 512, 0, stream>>>(attnS, af, gvec);
    {
        const long long total = (long long)N_ * 512;
        add_g_kernel<<<dim3((unsigned)((total + 255) / 256), 1, B_), 256, 0, stream>>>(af, gvec);
    }

    // ---- non-boundary MLP ----
    launch_gemm(npts, pw1, pb1, nb1, NN_, CH_, 64, true, stream);
    launch_gemm(nb1,  pw2, pb2, nb2, NN_, 64, 128, true, stream);
    launch_gemm(nb2,  pw3, pb3, nb3, NN_, 128, 512, true, stream);

    // ---- boundary edge branch ----
    bknn_kernel<<<dim3(NB_, 1, B_), 256, 0, stream>>>(ptsT, sq, bpts, bknn);
    edge_kernel<<<dim3((NB_ * KNN_ + 255) / 256, 1, B_), 256, 0, stream>>>(ptsT, bpts, bknn, edge);
    launch_gemm(edge, bw1, bb1, e1, NB_ * KNN_, 2 * CH_, 64, true, stream);
    launch_gemm(e1,   bw2, bb2, e2, NB_ * KNN_, 64, 128, true, stream);
    launch_gemm(e2,   bw3, bb3, e3, NB_ * KNN_, 128, 256, true, stream);
    maxpool_kernel<<<dim3((NB_ * 256 + 255) / 256, 1, B_), 256, 0, stream>>>(e3, bfeat);

    // ---- concats ----
    concat_b_kernel<<<dim3((NB_ * 768 + 255) / 256, 1, B_), 256, 0, stream>>>(bfeat, af, order, bcat);
    {
        const long long total = (long long)NN_ * 1024;
        concat_n_kernel<<<dim3((unsigned)((total + 255) / 256), 1, B_), 256, 0, stream>>>(nb3, af, order, ncat);
    }

    // ---- heads ----
    launch_gemm(bcat, c1, cb1, bh1, NB_, 768, 512, true, stream);
    launch_gemm(bh1,  c2, cb2, bh2, NB_, 512, 256, true, stream);
    launch_gemm(bh2,  c3, cb3, bh3, NB_, 256, 128, true, stream);
    launch_gemm(bh3,  c4, cb4, blog, NB_, 128, NCLS_, false, stream);
    launch_gemm(ncat, c5, cb5, nh1, NN_, 1024, 512, true, stream);
    launch_gemm(nh1,  c6, cb6, nh2, NN_, 512, 256, true, stream);
    launch_gemm(nh2,  c7, cb7, nh3, NN_, 256, 128, true, stream);
    launch_gemm(nh3,  c8, cb8, nlog, NN_, 128, NCLS_, false, stream);

    // ---- output ----
    logsoftmax_scatter_kernel<<<dim3((N_ + 255) / 256, 1, B_), 256, 0, stream>>>(blog, nlog, order, out);
}
